// CharRNN_21397527068870
// MI455X (gfx1250) — matrix operand order repaired and checked by reading, not compiled, not run on hardware
//
#include <hip/hip_runtime.h>
#include <hip/hip_bf16.h>

// ---------------- problem constants ----------------
constexpr int kV = 32000, kB = 64, kT = 50, kR = 1024, kU = 256;
constexpr int kCut = 2000, kTailP = 64;
constexpr int kNTail = kV - kCut;          // 30000
constexpr int kNHead = kCut + 1;           // 2001
constexpr int kNHeadP = 2016;              // padded to 16
constexpr int kRows = kB * kT;             // 3200

typedef __bf16 bf16;
typedef __attribute__((ext_vector_type(16))) __bf16 bf16x16;
typedef __attribute__((ext_vector_type(8)))  __bf16 bf16x8;
typedef __attribute__((ext_vector_type(8)))  float  f32x8;

__device__ __forceinline__ bf16x16 cat16(bf16x8 lo, bf16x8 hi) {
    return __builtin_shufflevector(lo, hi, 0,1,2,3,4,5,6,7,8,9,10,11,12,13,14,15);
}

// Load one A fragment (16x32 bf16, ISA 7.12.2 layout) for this lane.
// j=0..7 -> k = 8*half + j ; j=8..15 -> k = 16 + 8*half + (j-8)
__device__ __forceinline__ bf16x16 load_afrag(const bf16* __restrict__ Arow, int half) {
    bf16x8 a0 = *reinterpret_cast<const bf16x8*>(Arow + 8 * half);
    bf16x8 a1 = *reinterpret_cast<const bf16x8*>(Arow + 16 + 8 * half);
    return cat16(a0, a1);
}

// Load one B fragment (32x16 bf16) from transposed weight row.
// j=0..7 -> k = 8*half + j ; j=8..15 -> k = 16 + 8*half + (j-8)
__device__ __forceinline__ bf16x16 load_bfrag(const bf16* __restrict__ Bseg, int half) {
    bf16x8 b0 = *reinterpret_cast<const bf16x8*>(Bseg + 8 * half);
    bf16x8 b1 = *reinterpret_cast<const bf16x8*>(Bseg + 16 + 8 * half);
    return cat16(b0, b1);
}

// Affine K-span of 16x16x32 WMMAs: A row pointer fixed, offsets constant.
__device__ __forceinline__ f32x8 wmma_span(const bf16* __restrict__ Arow,
                                           const bf16* __restrict__ Bseg,
                                           int kn, int half, f32x8 acc) {
    for (int k0 = 0; k0 < kn; k0 += 32) {
        __builtin_prefetch(Bseg + k0 + 256, 0, 1);     // global_prefetch_b8
        bf16x16 a = load_afrag(Arow + k0, half);
        bf16x16 b = load_bfrag(Bseg + k0, half);
        acc = __builtin_amdgcn_wmma_f32_16x16x32_bf16(false, a, false, b,
                                                      (short)0, acc, false, false);
    }
    return acc;
}

// One 16x16 output tile. A row-major [M,K] split across two buffers at
// column Ksplit (multiple of 32). Bt transposed weight, row-major [N,Ktot].
__device__ __forceinline__ f32x8 wmma_tile(
    const bf16* __restrict__ Alo, int ldAlo,
    const bf16* __restrict__ Ahi, int ldAhi,
    int Ksplit, int Ktot,
    const bf16* __restrict__ Bt,
    int mt, int nt, int lane)
{
    const int half = lane >> 4;
    const int mrow = mt * 16 + (lane & 15);
    const int n    = nt * 16 + (lane & 15);
    const bf16* Brow = Bt + (size_t)n * Ktot;
    f32x8 acc = {0.f,0.f,0.f,0.f,0.f,0.f,0.f,0.f};
    acc = wmma_span(Alo + (size_t)mrow * ldAlo, Brow, Ksplit, half, acc);
    acc = wmma_span(Ahi + (size_t)mrow * ldAhi, Brow + Ksplit, Ktot - Ksplit, half, acc);
    return acc;
}

// ---------------- prep kernels ----------------
__global__ void zero_u32_kernel(unsigned* p, int n) {
    int i = blockIdx.x * blockDim.x + threadIdx.x;
    if (i < n) p[i] = 0u;
}

// src row-major [K,N] fp32 -> dst row-major [Npad,K] bf16 (zero pad rows)
__global__ void transpose_bf16_kernel(const float* __restrict__ src, int K, int N, int Npad,
                                      bf16* __restrict__ dst) {
    long i = (long)blockIdx.x * blockDim.x + threadIdx.x;
    long total = (long)Npad * K;
    if (i >= total) return;
    int n = (int)(i / K), k = (int)(i % K);
    dst[i] = (n < N) ? (bf16)src[(long)k * N + n] : (bf16)0.f;
}

__global__ void embed_kernel(const int* __restrict__ ids, const float* __restrict__ emb,
                             bf16* __restrict__ X) {
    long i = (long)blockIdx.x * blockDim.x + threadIdx.x;
    if (i >= (long)kRows * kU) return;
    int u = (int)(i % kU);
    long bt = i / kU;
    int t = (int)(bt % kT), b = (int)(bt / kT);
    int idx = ids[b * kT + t];
    X[((long)t * kB + b) * kU + u] = (bf16)emb[(long)idx * kU + u];
}

// ---------------- GRU step kernels ----------------
// gates: g = sigmoid([x,h] @ Wg + bg), N = 2R. n<R -> write r*h (bf16); n>=R -> write u (f32)
__global__ __launch_bounds__(256)
void gru_gates_kernel(const bf16* Alo, int ldAlo, const bf16* Ahi, int ldAhi,
                      int Ksplit, int Ktot, const bf16* Wt, const float* bias,
                      const float* hf, bf16* rh_out, float* u_out) {
    int lane = threadIdx.x & 31, wave = threadIdx.x >> 5;
    int tile = blockIdx.x * (blockDim.x >> 5) + wave;
    int mt = tile & 3, nt = tile >> 2;          // 4 x 128 tiles
    f32x8 acc = wmma_tile(Alo, ldAlo, Ahi, ldAhi, Ksplit, Ktot, Wt, mt, nt, lane);
    union { f32x8 v; float f[8]; } d; d.v = acc;
    int half = lane >> 4;
    int n = nt * 16 + (lane & 15);
    float b = bias[n];
#pragma unroll
    for (int r = 0; r < 8; ++r) {
        int m = mt * 16 + r + 8 * half;
        float g = 1.f / (1.f + __expf(-(d.f[r] + b)));
        if (n < kR) rh_out[(size_t)m * kR + n] = (bf16)(g * hf[(size_t)m * kR + n]);
        else        u_out[(size_t)m * kR + (n - kR)] = g;
    }
}

// candidate: c = tanh([x, r*h] @ Wc + bc); h' = u*h + (1-u)*c  (N = R)
__global__ __launch_bounds__(256)
void gru_cand_kernel(const bf16* Alo, int ldAlo, const bf16* Ahi, int ldAhi,
                     int Ksplit, int Ktot, const bf16* Wt, const float* bias,
                     float* hf, const float* u, bf16* hbf) {
    int lane = threadIdx.x & 31, wave = threadIdx.x >> 5;
    int tile = blockIdx.x * (blockDim.x >> 5) + wave;
    int mt = tile & 3, nt = tile >> 2;          // 4 x 64 tiles
    f32x8 acc = wmma_tile(Alo, ldAlo, Ahi, ldAhi, Ksplit, Ktot, Wt, mt, nt, lane);
    union { f32x8 v; float f[8]; } d; d.v = acc;
    int half = lane >> 4;
    int n = nt * 16 + (lane & 15);
    float b = bias[n];
#pragma unroll
    for (int r = 0; r < 8; ++r) {
        int m = mt * 16 + r + 8 * half;
        size_t ix = (size_t)m * kR + n;
        float c = tanhf(d.f[r] + b);
        float uu = u[ix];
        float hn = uu * hf[ix] + (1.f - uu) * c;
        hf[ix] = hn;
        hbf[ix] = (bf16)hn;
    }
}

// projection: outs[(b*T + t), :] = h2 @ Wp + bp  (N = U)
__global__ __launch_bounds__(256)
void proj_kernel(const bf16* A, const bf16* Wt, const float* bias, int t, bf16* outs) {
    int lane = threadIdx.x & 31, wave = threadIdx.x >> 5;
    int tile = blockIdx.x * (blockDim.x >> 5) + wave;
    int mt = tile & 3, nt = tile >> 2;          // 4 x 16 tiles
    f32x8 acc = wmma_tile(A, kR, A, kR, kR, kR, Wt, mt, nt, lane);
    union { f32x8 v; float f[8]; } d; d.v = acc;
    int half = lane >> 4;
    int n = nt * 16 + (lane & 15);
    float b = bias[n];
#pragma unroll
    for (int r = 0; r < 8; ++r) {
        int m = mt * 16 + r + 8 * half;                 // batch index
        outs[((size_t)m * kT + t) * kU + n] = (bf16)(d.f[r] + b);
    }
}

// generic GEMM -> bf16 (tail low-rank projection: [3200,256] x [256,64])
__global__ __launch_bounds__(256)
void gemm_bf16_kernel(const bf16* A, int K, const bf16* Wt, int Ntiles, int N, bf16* out) {
    int lane = threadIdx.x & 31, wave = threadIdx.x >> 5;
    int tile = blockIdx.x * (blockDim.x >> 5) + wave;
    int mt = tile / Ntiles, nt = tile % Ntiles;
    f32x8 acc = wmma_tile(A, K, A, K, K, K, Wt, mt, nt, lane);
    union { f32x8 v; float f[8]; } d; d.v = acc;
    int half = lane >> 4;
    int n = nt * 16 + (lane & 15);
#pragma unroll
    for (int r = 0; r < 8; ++r) {
        int m = mt * 16 + r + 8 * half;
        out[(size_t)m * N + n] = (bf16)d.f[r];
    }
}

// Fused GEMM + online logsumexp + target gather -> per-row loss.
// KC: compile-time K so the 16xKC A tile is cached in registers once.
// MODE 0 (head): losses[row]  = lse - logit[min(lbl,cut)]
// MODE 1 (tail): losses[row] += (lbl>=cut) ? lse - logit[clip(lbl-cut)] : 0
template<int KC, int MODE>
__global__ __launch_bounds__(256)
void softmax_loss_kernel(const bf16* __restrict__ A, const bf16* __restrict__ Wt,
                         int Ntiles, int Nvalid,
                         const int* __restrict__ labels, int cut,
                         float* __restrict__ losses) {
    constexpr int KF = KC / 32;                 // A fragments along K
    __shared__ float sm[8][16], ss[8][16], st[8][16];
    int lane = threadIdx.x & 31, wave = threadIdx.x >> 5;
    int nwaves = blockDim.x >> 5;
    int rowbase = blockIdx.x * 16;
    int half = lane >> 4;
    int col = lane & 15;

    // cache this block's A tile in registers (shared by every N tile)
    const bf16* Arow = A + (size_t)(rowbase + col) * KC;
    bf16x16 afrag[KF];
#pragma unroll
    for (int kc = 0; kc < KF; ++kc) afrag[kc] = load_afrag(Arow + kc * 32, half);

    int tgt[8];
#pragma unroll
    for (int r = 0; r < 8; ++r) {
        int lbl = labels[rowbase + r + 8 * half];
        if (MODE == 0) tgt[r] = (lbl < cut) ? lbl : cut;
        else { int c = lbl - cut; if (c < 0) c = 0; if (c > Nvalid - 1) c = Nvalid - 1; tgt[r] = c; }
    }
    float run_m[8], run_s[8], tl[8];
#pragma unroll
    for (int r = 0; r < 8; ++r) { run_m[r] = -1e30f; run_s[r] = 0.f; tl[r] = -1e30f; }

    for (int nt = wave; nt < Ntiles; nt += nwaves) {
        int n = nt * 16 + col;
        const bf16* Brow = Wt + (size_t)n * KC;
        f32x8 acc = {0.f,0.f,0.f,0.f,0.f,0.f,0.f,0.f};
#pragma unroll
        for (int kc = 0; kc < KF; ++kc) {
            bf16x16 b = load_bfrag(Brow + kc * 32, half);
            acc = __builtin_amdgcn_wmma_f32_16x16x32_bf16(false, afrag[kc], false, b,
                                                          (short)0, acc, false, false);
        }
        union { f32x8 v; float f[8]; } d; d.v = acc;
        bool valid = (n < Nvalid);
#pragma unroll
        for (int r = 0; r < 8; ++r) {
            float x = valid ? d.f[r] : -1e30f;
            if (valid && n == tgt[r]) tl[r] = d.f[r];
            float nm = fmaxf(run_m[r], x);
            float e  = valid ? __expf(x - nm) : 0.f;
            run_s[r] = run_s[r] * __expf(run_m[r] - nm) + e;
            run_m[r] = nm;
        }
    }
    // reduce across the 16 lanes of each half-wave (columns of the tile)
#pragma unroll
    for (int mask = 1; mask < 16; mask <<= 1) {
#pragma unroll
        for (int r = 0; r < 8; ++r) {
            float om = __shfl_xor(run_m[r], mask, 32);
            float os = __shfl_xor(run_s[r], mask, 32);
            float ot = __shfl_xor(tl[r],    mask, 32);
            float nm = fmaxf(run_m[r], om);
            run_s[r] = run_s[r] * __expf(run_m[r] - nm) + os * __expf(om - nm);
            run_m[r] = nm;
            tl[r] = fmaxf(tl[r], ot);
        }
    }
    if (col == 0) {
#pragma unroll
        for (int r = 0; r < 8; ++r) {
            int rl = r + 8 * half;
            sm[wave][rl] = run_m[r]; ss[wave][rl] = run_s[r]; st[wave][rl] = tl[r];
        }
    }
    __syncthreads();
    if (threadIdx.x < 16) {
        float M = -1e30f, S = 0.f, TT = -1e30f;
        for (int w = 0; w < nwaves; ++w) {
            float om = sm[w][threadIdx.x], os = ss[w][threadIdx.x];
            float nm = fmaxf(M, om);
            S = S * __expf(M - nm) + os * __expf(om - nm);
            M = nm;
            TT = fmaxf(TT, st[w][threadIdx.x]);
        }
        int row = rowbase + threadIdx.x;
        float loss = (M + __logf(S)) - TT;
        if (MODE == 1) {
            if (labels[row] < cut) loss = 0.f;
            losses[row] += loss;
        } else {
            losses[row] = loss;
        }
    }
}

__global__ void reduce_mean_kernel(const float* losses, int n, float* out) {
    __shared__ float s[256];
    float a = 0.f;
    for (int i = threadIdx.x; i < n; i += blockDim.x) a += losses[i];
    s[threadIdx.x] = a; __syncthreads();
    for (int stp = 128; stp > 0; stp >>= 1) {
        if ((int)threadIdx.x < stp) s[threadIdx.x] += s[threadIdx.x + stp];
        __syncthreads();
    }
    if (threadIdx.x == 0) out[0] = s[0] / (float)n;
}

// ---------------- host ----------------
extern "C" void kernel_launch(void* const* d_in, const int* in_sizes, int n_in,
                              void* d_out, int out_size, void* d_ws, size_t ws_size,
                              hipStream_t stream) {
    const int*   ids    = (const int*)  d_in[0];
    const int*   tgts   = (const int*)  d_in[1];
    const float* emb    = (const float*)d_in[2];
    const float* Wg1 = (const float*)d_in[3];  const float* bg1 = (const float*)d_in[4];
    const float* Wc1 = (const float*)d_in[5];  const float* bc1 = (const float*)d_in[6];
    const float* Wg2 = (const float*)d_in[7];  const float* bg2 = (const float*)d_in[8];
    const float* Wc2 = (const float*)d_in[9];  const float* bc2 = (const float*)d_in[10];
    const float* Wp  = (const float*)d_in[11]; const float* bp  = (const float*)d_in[12];
    const float* Whd = (const float*)d_in[13];
    const float* Wtp = (const float*)d_in[14];
    const float* Wtl = (const float*)d_in[15];
    float* out = (float*)d_out;

    char* p = (char*)d_ws;
    auto carve = [&](size_t bytes) -> char* {
        char* r = p; p += (bytes + 255) & ~(size_t)255; return r;
    };
    // zero-init block (contiguous, every size a multiple of 256B)
    float* h1f = (float*)carve((size_t)kB * kR * 4);
    float* h2f = (float*)carve((size_t)kB * kR * 4);
    bf16* h1bf = (bf16*)carve((size_t)kB * kR * 2);
    bf16* h2bf = (bf16*)carve((size_t)kB * kR * 2);
    float* losses = (float*)carve((size_t)kRows * 4);
    size_t zero_bytes = (char*)p - (char*)h1f;
    // remaining scratch
    float* u1  = (float*)carve((size_t)kB * kR * 4);
    float* u2  = (float*)carve((size_t)kB * kR * 4);
    bf16* rh1  = (bf16*)carve((size_t)kB * kR * 2);
    bf16* rh2  = (bf16*)carve((size_t)kB * kR * 2);
    bf16* Xemb = (bf16*)carve((size_t)kT * kB * kU * 2);
    bf16* outs = (bf16*)carve((size_t)kRows * kU * 2);
    bf16* ptail = (bf16*)carve((size_t)kRows * kTailP * 2);
    bf16* Wg1t = (bf16*)carve((size_t)(2 * kR) * (kU + kR) * 2);
    bf16* Wc1t = (bf16*)carve((size_t)kR * (kU + kR) * 2);
    bf16* Wg2t = (bf16*)carve((size_t)(2 * kR) * (2 * kR) * 2);
    bf16* Wc2t = (bf16*)carve((size_t)kR * (2 * kR) * 2);
    bf16* Wpt  = (bf16*)carve((size_t)kU * kR * 2);
    bf16* Whdt = (bf16*)carve((size_t)kNHeadP * kU * 2);
    bf16* Wtpt = (bf16*)carve((size_t)kTailP * kU * 2);
    bf16* Wtlt = (bf16*)carve((size_t)kNTail * kTailP * 2);

    // state/loss zero init
    {
        int n = (int)(zero_bytes / 4);
        zero_u32_kernel<<<(n + 255) / 256, 256, 0, stream>>>((unsigned*)h1f, n);
    }
    // weight convert + transpose (one-time, fp32 -> bf16 [N,K])
    auto tr = [&](const float* src, int K, int N, int Npad, bf16* dst) {
        long tot = (long)Npad * K;
        transpose_bf16_kernel<<<(unsigned)((tot + 255) / 256), 256, 0, stream>>>(src, K, N, Npad, dst);
    };
    tr(Wg1, kU + kR, 2 * kR, 2 * kR, Wg1t);
    tr(Wc1, kU + kR, kR,     kR,     Wc1t);
    tr(Wg2, 2 * kR,  2 * kR, 2 * kR, Wg2t);
    tr(Wc2, 2 * kR,  kR,     kR,     Wc2t);
    tr(Wp,  kR,      kU,     kU,     Wpt);
    tr(Whd, kU,      kNHead, kNHeadP, Whdt);
    tr(Wtp, kU,      kTailP, kTailP, Wtpt);
    tr(Wtl, kTailP,  kNTail, kNTail, Wtlt);
    // embedding gather -> bf16 [T,B,U]
    {
        long tot = (long)kRows * kU;
        embed_kernel<<<(unsigned)((tot + 255) / 256), 256, 0, stream>>>(ids, emb, Xemb);
    }
    // recurrence: 50 sequential steps, weights L2-resident after first pass
    for (int t = 0; t < kT; ++t) {
        const bf16* Xt = Xemb + (size_t)t * kB * kU;
        gru_gates_kernel<<<64, 256, 0, stream>>>(Xt, kU, h1bf, kR, kU, kU + kR,
                                                 Wg1t, bg1, h1f, rh1, u1);
        gru_cand_kernel<<<32, 256, 0, stream>>>(Xt, kU, rh1, kR, kU, kU + kR,
                                                Wc1t, bc1, h1f, u1, h1bf);
        gru_gates_kernel<<<64, 256, 0, stream>>>(h1bf, kR, h2bf, kR, kR, 2 * kR,
                                                 Wg2t, bg2, h2f, rh2, u2);
        gru_cand_kernel<<<32, 256, 0, stream>>>(h1bf, kR, rh2, kR, kR, 2 * kR,
                                                Wc2t, bc2, h2f, u2, h2bf);
        proj_kernel<<<8, 256, 0, stream>>>(h2bf, Wpt, bp, t, outs);
    }
    // tail low-rank projection: [3200,256] x [256,64]
    gemm_bf16_kernel<<<100, 256, 0, stream>>>(outs, kU, Wtpt, kTailP / 16, kTailP, ptail);
    // head loss (N=2001 padded to 2016, masked) and tail loss (N=30000), fused online LSE
    softmax_loss_kernel<kU, 0><<<kRows / 16, 256, 0, stream>>>(outs, Whdt, kNHeadP / 16,
                                                               kNHead, tgts, kCut, losses);
    softmax_loss_kernel<kTailP, 1><<<kRows / 16, 256, 0, stream>>>(ptail, Wtlt, kNTail / 16,
                                                                   kNTail, tgts, kCut, losses);
    reduce_mean_kernel<<<1, 256, 0, stream>>>(losses, kRows, out);
    (void)in_sizes; (void)n_in; (void)out_size; (void)ws_size;
}
